// InfoNCELoss_47906065219603
// MI455X (gfx1250) — matrix operand order repaired and checked by reading, not compiled, not run on hardware
//
#include <hip/hip_runtime.h>
#include <math.h>

// ---- CDNA5 WMMA types ----
typedef __attribute__((ext_vector_type(16))) _Float16 v16h;
typedef __attribute__((ext_vector_type(8)))  _Float16 v8h;
typedef __attribute__((ext_vector_type(8)))  float    v8f;

#define BSZ 8192
#define DSZ 256
#define TEMP 0.07f
#define EPSL 1e-8f
#define NLN2 -0.6931471805599453f   // -ln(2)

#define LDP 264                      // padded LDS row length in halves (+16B pad)
#define LDS_BUF_BYTES (32 * LDP * 2) // 16896 bytes per buffer

#define ACC_WORDS 18

// Kernel A: zero accumulators + histogram.
__global__ void supcon_init(float* __restrict__ acc) {
  const int i = threadIdx.x;
  if (i < ACC_WORDS) acc[i] = 0.0f;
}

// Kernel B: f32 -> f16 with sqrt(log2(e)/T) folded in, so exp(sim/T)==exp2(dot).
// Also build the 16-bin label histogram (num_pos/num_neg per row for free).
__global__ __launch_bounds__(256) void supcon_convert(
    const float* __restrict__ f, const int* __restrict__ labels,
    _Float16* __restrict__ fh, float scale, float* __restrict__ acc) {
  const int i = blockIdx.x * blockDim.x + threadIdx.x;
  if (i < BSZ * DSZ) fh[i] = (_Float16)(f[i] * scale);
  if (i < BSZ) atomicAdd(&acc[2 + labels[i]], 1.0f);
}

// Main kernel: 4 waves per block, each wave owns a 16-row block with its A
// fragments register-resident. B tiles (32 columns x 256 k) are staged into
// double-buffered LDS with global_load_async_to_lds_b128 and shared by all 4
// waves (4x less L2 traffic than per-wave global streaming).
__global__ __launch_bounds__(128) void supcon_main(
    const _Float16* __restrict__ Fh, const int* __restrict__ labels,
    const float* __restrict__ acc_ro, float* __restrict__ acc) {
  __shared__ _Float16 ldsb[2][32][LDP];

  const int tid  = threadIdx.x;
  const int lane = tid & 31;
  const int wv   = tid >> 5;
  const int hsel = lane >> 4;
  const int l15  = lane & 15;
  const int m0   = blockIdx.x * 64 + wv * 16;

  // ---- async-copy mapping: 128 threads cover 32 rows x 512B; each thread
  // copies 128 bytes as 8 x b128 chunks (INST_OFFSET hits LDS and global).
  const int crow = tid >> 2;   // 0..31 (B row within tile)
  const int cq   = tid & 3;    // 0..3  (128B quarter of the row)
  const uint32_t ldsDst0 =
      (uint32_t)(uintptr_t)(&ldsb[0][0][0]) + (uint32_t)(crow * (LDP * 2) + cq * 128);
  const uint64_t gThread =
      (uint64_t)(uintptr_t)Fh + (uint64_t)crow * (DSZ * 2) + (uint64_t)cq * 128;

  auto issue_async = [&](int bufsel, int n0v) {
    const uint32_t dst = ldsDst0 + (uint32_t)bufsel * LDS_BUF_BYTES;
    const uint64_t ga  = gThread + (uint64_t)n0v * (DSZ * 2);
    asm volatile(
        "global_load_async_to_lds_b128 %0, %1, off\n\t"
        "global_load_async_to_lds_b128 %0, %1, off offset:16\n\t"
        "global_load_async_to_lds_b128 %0, %1, off offset:32\n\t"
        "global_load_async_to_lds_b128 %0, %1, off offset:48\n\t"
        "global_load_async_to_lds_b128 %0, %1, off offset:64\n\t"
        "global_load_async_to_lds_b128 %0, %1, off offset:80\n\t"
        "global_load_async_to_lds_b128 %0, %1, off offset:96\n\t"
        "global_load_async_to_lds_b128 %0, %1, off offset:112"
        :: "v"(dst), "v"(ga) : "memory");
  };

  // ---- Preload A fragments: contiguous k order, k = kc*32 + 16*hsel + s ----
  v16h a[8];
  {
    const _Float16* rowp = Fh + (size_t)(m0 + l15) * DSZ;
    #pragma unroll
    for (int kc = 0; kc < 8; ++kc) {
      const int kb = kc * 32 + hsel * 16;
      v8h lo = *(const v8h*)(rowp + kb);
      v8h hi = *(const v8h*)(rowp + kb + 8);
      v16h t;
      #pragma unroll
      for (int e = 0; e < 8; ++e) { t[e] = lo[e]; t[e + 8] = hi[e]; }
      a[kc] = t;
    }
  }

  // C/D layout: VGPR r holds row (rbase + r), column (n0 + l15)
  const int rbase = m0 + hsel * 8;
  int   lr[8];
  float hp[8];
  #pragma unroll
  for (int r = 0; r < 8; ++r) {
    lr[r] = labels[rbase + r];
    hp[r] = acc_ro[2 + lr[r]];
  }

  float negAcc[8], lossAcc[8], negRow[8];
  #pragma unroll
  for (int r = 0; r < 8; ++r) { negAcc[r] = 0.f; lossAcc[r] = 0.f; negRow[r] = 0.f; }

  issue_async(0, 0);
  int buf = 0;

  const int NIT = 2 * (BSZ / 32);          // pass1 tiles then pass2 tiles
  for (int it = 0; it < NIT; ++it) {
    const int  n0    = (it * 32) & (BSZ - 1);
    const bool pass2 = (it >= BSZ / 32);

    issue_async(buf ^ 1, (n0 + 32) & (BSZ - 1));     // prefetch next buffer
    asm volatile("s_wait_asynccnt 0x8" ::: "memory"); // my current-buf writes done
    __syncthreads();                                  // everyone's writes visible

    const int nA = n0 + l15, nB = nA + 16;
    const int lcA = labels[nA], lcB = labels[nB];
    const _Float16* bA = &ldsb[buf][l15][hsel * 16];
    const _Float16* bB = &ldsb[buf][16 + l15][hsel * 16];

    v8f c0 = {}, c1 = {};
    #pragma unroll
    for (int kc = 0; kc < 8; ++kc) {
      v8h a0 = *(const v8h*)(bA + kc * 32);
      v8h a1 = *(const v8h*)(bA + kc * 32 + 8);
      v8h b0l = *(const v8h*)(bB + kc * 32);
      v8h b1l = *(const v8h*)(bB + kc * 32 + 8);
      v16h bfA, bfB;
      #pragma unroll
      for (int e = 0; e < 8; ++e) {
        bfA[e] = a0[e]; bfA[e + 8] = a1[e];
        bfB[e] = b0l[e]; bfB[e + 8] = b1l[e];
      }
      c0 = __builtin_amdgcn_wmma_f32_16x16x32_f16(false, a[kc], false, bfA, (short)0, c0, false, false);
      c1 = __builtin_amdgcn_wmma_f32_16x16x32_f16(false, a[kc], false, bfB, (short)0, c1, false, false);
    }

    if (!pass2) {
      #pragma unroll
      for (int r = 0; r < 8; ++r) {
        // raw v_exp_f32: inputs bounded (|x| <= ~20.6), no denorm guard needed
        const float e0 = __builtin_amdgcn_exp2f(c0[r]);   // == exp(sim/T)
        const float e1 = __builtin_amdgcn_exp2f(c1[r]);
        negAcc[r] += (lr[r] == lcA) ? 0.f : e0;
        negAcc[r] += (lr[r] == lcB) ? 0.f : e1;
      }
    } else {
      #pragma unroll
      for (int r = 0; r < 8; ++r) {
        const int   m   = rbase + r;
        const float e0  = __builtin_amdgcn_exp2f(c0[r]);
        const float e1  = __builtin_amdgcn_exp2f(c1[r]);
        const float S   = negRow[r];
        // ratio = e/(e+S) via raw v_rcp_f32; pair loss = -ln(ratio+eps)
        const float r0  = e0 * __builtin_amdgcn_rcpf(e0 + S);
        const float r1  = e1 * __builtin_amdgcn_rcpf(e1 + S);
        const float pl0 = __builtin_amdgcn_logf(r0 + EPSL) * NLN2; // -ln(x)
        const float pl1 = __builtin_amdgcn_logf(r1 + EPSL) * NLN2;
        const bool  p0  = (lr[r] == lcA) & (m != nA);
        const bool  p1  = (lr[r] == lcB) & (m != nB);
        lossAcc[r] += p0 ? pl0 : 0.f;
        lossAcc[r] += p1 ? pl1 : 0.f;
      }
    }

    __syncthreads();          // all reads of buf done before it is rewritten
    buf ^= 1;

    if (it == (BSZ / 32) - 1) {
      // pass-1 done: reduce neg sums across the 16 lanes of each half
      #pragma unroll
      for (int r = 0; r < 8; ++r) {
        float v = negAcc[r];
        #pragma unroll
        for (int msk = 1; msk < 16; msk <<= 1) v += __shfl_xor(v, msk, 32);
        negRow[r] = v;
      }
    }
  }

  #pragma unroll
  for (int r = 0; r < 8; ++r) {
    float v = lossAcc[r];
    #pragma unroll
    for (int msk = 1; msk < 16; msk <<= 1) v += __shfl_xor(v, msk, 32);
    lossAcc[r] = v;
  }

  if (l15 == 0) {
    float tot = 0.f, cnt = 0.f;
    #pragma unroll
    for (int r = 0; r < 8; ++r) {
      const float numPos = hp[r] - 1.0f;
      const float numNeg = (float)BSZ - hp[r];
      const bool  valid  = (numPos > 0.f) && (numNeg > 0.f);
      const float rl     = lossAcc[r] / fmaxf(numPos, 1.f);
      tot += valid ? rl : 0.f;
      cnt += valid ? 1.f : 0.f;
    }
    atomicAdd(&acc[0], tot);
    atomicAdd(&acc[1], cnt);
  }
}

__global__ void supcon_finalize(const float* __restrict__ acc,
                                float* __restrict__ out) {
  if (threadIdx.x == 0 && blockIdx.x == 0) {
    const float cnt = acc[1];
    out[0] = (cnt > 0.f) ? acc[0] / fmaxf(cnt, 1.f) : 0.0f;
  }
}

extern "C" void kernel_launch(void* const* d_in, const int* in_sizes, int n_in,
                              void* d_out, int out_size, void* d_ws, size_t ws_size,
                              hipStream_t stream) {
  const float* feats  = (const float*)d_in[0];
  const int*   labels = (const int*)d_in[1];

  _Float16* Fh  = (_Float16*)d_ws;                                  // 4 MB
  float*    acc = (float*)((char*)d_ws + (size_t)BSZ * DSZ * sizeof(_Float16));

  const float scale = sqrtf(1.44269504089f / TEMP);  // sqrt(log2e / T)

  supcon_init<<<1, 32, 0, stream>>>(acc);
  supcon_convert<<<(BSZ * DSZ + 255) / 256, 256, 0, stream>>>(feats, labels, Fh, scale, acc);
  supcon_main<<<BSZ / 64, 128, 0, stream>>>(Fh, labels, acc, acc);
  supcon_finalize<<<1, 32, 0, stream>>>(acc, (float*)d_out);
}
